// Attention_64312840290425
// MI455X (gfx1250) — matrix operand order repaired and checked by reading, not compiled, not run on hardware
//
#include <hip/hip_runtime.h>
#include <hip/hip_bf16.h>
#include <stdint.h>

typedef __attribute__((ext_vector_type(16))) __bf16 v16bf;
typedef __attribute__((ext_vector_type(8)))  __bf16 v8bf;
typedef __attribute__((ext_vector_type(4)))  __bf16 v4bf;
typedef __attribute__((ext_vector_type(8)))  float  v8f;
typedef __attribute__((vector_size(16)))     int    i32x4;

#define LOG2E 1.44269504088896340736f

static __device__ __forceinline__ __bf16 f2bf(float f) { return (__bf16)f; }

static __device__ __forceinline__ v16bf cat16(v8bf lo, v8bf hi) {
  return __builtin_shufflevector(lo, hi, 0,1,2,3,4,5,6,7,8,9,10,11,12,13,14,15);
}

// CDNA5 async global->LDS copy (GLOBAL_LOAD_ASYNC_TO_LDS_B128, ASYNCcnt-tracked).
// AS3 pointer representation == low 32 bits of the generic LDS address (ISA 10.2).
#if __has_builtin(__builtin_amdgcn_global_load_async_to_lds_b128)
#define HAVE_ASYNC_LDS 1
#define ASYNC_COPY_B128(gsrc, ldst)                                           \
  __builtin_amdgcn_global_load_async_to_lds_b128(                             \
      (__attribute__((address_space(1))) i32x4*)(uintptr_t)(gsrc),            \
      (__attribute__((address_space(3))) i32x4*)(unsigned)(uintptr_t)(ldst),  \
      0, 0)
// Deferred wait: ASYNCcnt completes in order, so <=2 means "previous tile's
// two copies have landed; the next tile's two may still be in flight".
#define ASYNC_WAIT_LE0() asm volatile("s_wait_asynccnt 0" ::: "memory")
#define ASYNC_WAIT_LE2() asm volatile("s_wait_asynccnt 2" ::: "memory")
#else
#define HAVE_ASYNC_LDS 0
#define ASYNC_WAIT_LE0()
#define ASYNC_WAIT_LE2()
#endif

// ---------------------------------------------------------------------------
// Kernel 0: f32 -> bf16 conversion (weights)
// ---------------------------------------------------------------------------
__global__ void cvt_f32_bf16(const float4* __restrict__ in, __bf16* __restrict__ out, int n4) {
  int i = blockIdx.x * blockDim.x + threadIdx.x;
  if (i < n4) {
    float4 f = in[i];
    v4bf o;
    o[0] = f2bf(f.x); o[1] = f2bf(f.y); o[2] = f2bf(f.z); o[3] = f2bf(f.w);
    *(v4bf*)&out[(size_t)i * 4] = o;
  }
}

// ---------------------------------------------------------------------------
// Kernel 1: QKV projections.  C[8192,1024] = A[8192,2048](f32) x W[2048,1024](bf16)
// Output stored head-split: [B=4][H=16][T=2048][64] bf16.
// Block tile 64x64, 4 waves each 32x32, K-step 32, register-prefetch pipeline.
// ---------------------------------------------------------------------------
__global__ __launch_bounds__(128) void proj_gemm(
    const float* __restrict__ Qs, const float* __restrict__ Ks, const float* __restrict__ Vs,
    const __bf16* __restrict__ Wq, const __bf16* __restrict__ Wk, const __bf16* __restrict__ Wv,
    __bf16* __restrict__ qo, __bf16* __restrict__ ko, __bf16* __restrict__ vo) {
  const float* A; const __bf16* W; __bf16* D;
  if (blockIdx.z == 0)      { A = Qs; W = Wq; D = qo; }
  else if (blockIdx.z == 1) { A = Ks; W = Wk; D = ko; }
  else                      { A = Vs; W = Wv; D = vo; }
  const int KDIM = 2048, NDIM = 1024;

  __shared__ __align__(16) __bf16 Alds[64 * 32];   // [row][k] row-major
  __shared__ __align__(16) __bf16 Blds[64 * 32];   // [n][k]   (transposed W tile)

  int tid = threadIdx.x;
  int wid = tid >> 5, lane = tid & 31;
  int half = lane >> 4, nn = lane & 15;
  int bm = blockIdx.y * 64, bn = blockIdx.x * 64;
  int wm = (wid >> 1) * 32, wn = (wid & 1) * 32;

  int rowA = tid >> 1, cbA = (tid & 1) * 16;   // A-tile loader coords
  int kkB = tid >> 2, c0B = (tid & 3) * 16;    // B-tile loader coords

  float fA[16];
  v8bf wB0, wB1;
  auto loadA = [&](int k0) {
    const float4* src = (const float4*)(A + (size_t)(bm + rowA) * KDIM + k0 + cbA);
    *(float4*)(fA + 0)  = src[0];
    *(float4*)(fA + 4)  = src[1];
    *(float4*)(fA + 8)  = src[2];
    *(float4*)(fA + 12) = src[3];
  };
  auto loadB = [&](int k0) {
    const v8bf* src = (const v8bf*)(W + (size_t)(k0 + kkB) * NDIM + bn + c0B);
    wB0 = src[0]; wB1 = src[1];
  };

  loadA(0); loadB(0);
  v8f acc[2][2] = {};

  for (int k0 = 0; k0 < KDIM; k0 += 32) {
    __syncthreads();                 // all waves done reading previous tiles
    { // commit A tile (f32 -> bf16 via v_cvt_pk_bf16_f32)
      v8bf o0, o1;
#pragma unroll
      for (int j = 0; j < 8; j++) { o0[j] = f2bf(fA[j]); o1[j] = f2bf(fA[8 + j]); }
      *(v8bf*)&Alds[rowA * 32 + cbA]     = o0;
      *(v8bf*)&Alds[rowA * 32 + cbA + 8] = o1;
    }
    { // commit B tile transposed: W[k][n] -> Blds[n][k]
#pragma unroll
      for (int j = 0; j < 8; j++) {
        Blds[(c0B + j) * 32 + kkB]     = wB0[j];
        Blds[(c0B + 8 + j) * 32 + kkB] = wB1[j];
      }
    }
    __syncthreads();
    if (k0 + 32 < KDIM) { loadA(k0 + 32); loadB(k0 + 32); }  // prefetch next tile

    v16bf af[2], bfr[2];
#pragma unroll
    for (int i = 0; i < 2; i++) {
      int r = wm + i * 16 + nn;   // A-frag: lane holds row r, K = grp*16 + half*8 + j
      af[i] = cat16(*(const v8bf*)&Alds[r * 32 + half * 8],
                    *(const v8bf*)&Alds[r * 32 + 16 + half * 8]);
    }
#pragma unroll
    for (int j = 0; j < 2; j++) {
      int c = wn + j * 16 + nn;   // B-frag: lane holds col c, K = grp*16 + half*8 + j
      bfr[j] = cat16(*(const v8bf*)&Blds[c * 32 + half * 8],
                     *(const v8bf*)&Blds[c * 32 + 16 + half * 8]);
    }
#pragma unroll
    for (int i = 0; i < 2; i++)
#pragma unroll
      for (int j = 0; j < 2; j++)
        acc[i][j] = __builtin_amdgcn_wmma_f32_16x16x32_bf16(
            false, af[i], false, bfr[j], (short)0, acc[i][j], false, false);
  }

  // C store with head-split layout (b, h constant per block)
  {
    int b = bm >> 11, t0 = (bm & 2047) + wm, h = bn >> 6;
    __bf16* Dbase = D + ((size_t)((b << 4) + h) * 2048 + t0) * 64 + wn;
#pragma unroll
    for (int i = 0; i < 2; i++)
#pragma unroll
      for (int j = 0; j < 2; j++)
#pragma unroll
        for (int r = 0; r < 8; r++)
          Dbase[(size_t)(i * 16 + half * 8 + r) * 64 + j * 16 + nn] = f2bf(acc[i][j][r]);
  }
}

// ---------------------------------------------------------------------------
// Kernel 2: flash attention.  One WG = (b, h, 64-query block); 4 waves x 16 rows.
// K tiles: async DMA to LDS, double-buffered with deferred asynccnt waits.
// V tiles: register-prefetched, transpose-stored to LDS. Online softmax in f32.
// ---------------------------------------------------------------------------
__global__ __launch_bounds__(128) void flash_attn(
    const __bf16* __restrict__ q, const __bf16* __restrict__ k,
    const __bf16* __restrict__ v, __bf16* __restrict__ o) {
  const int T = 2048;
  const int NIT = T / 32;
  __shared__ __align__(16) __bf16 Klds[2][32 * 64];   // [buf][key][d]
  __shared__ __align__(16) __bf16 Vlds[64 * 32];      // [d][key]  (transposed)
  __shared__ __align__(16) __bf16 Plds[4][16 * 32];   // per-wave P tile [row][key]

  int tid = threadIdx.x, wid = tid >> 5, lane = tid & 31;
  int half = lane >> 4, nn = lane & 15;
  int gid = blockIdx.x;
  int qb = gid & 31;            // T/64 = 32 query blocks
  int h  = (gid >> 5) & 15;
  int b  = gid >> 9;

  const __bf16* qpanel = q + ((size_t)(b * 16 + h) * T + qb * 64 + wid * 16) * 64;
  const __bf16* kpanel = k + (size_t)(b * 16 + h) * T * 64;
  const __bf16* vpanel = v + (size_t)(b * 16 + h) * T * 64;

  int key = tid >> 2, c0 = (tid & 3) * 16;   // KV loader coords

  auto stageK = [&](int kv0, int buf) {
    const __bf16* gk = kpanel + (size_t)(kv0 + key) * 64 + c0;
#if HAVE_ASYNC_LDS
    ASYNC_COPY_B128(gk,     &Klds[buf][key * 64 + c0]);
    ASYNC_COPY_B128(gk + 8, &Klds[buf][key * 64 + c0 + 8]);
#else
    const v8bf* s = (const v8bf*)gk;
    *(v8bf*)&Klds[buf][key * 64 + c0]     = s[0];
    *(v8bf*)&Klds[buf][key * 64 + c0 + 8] = s[1];
#endif
  };
  v8bf vr0, vr1;
  auto loadV = [&](int kv0) {
    const v8bf* vs = (const v8bf*)(vpanel + (size_t)(kv0 + key) * 64 + c0);
    vr0 = vs[0]; vr1 = vs[1];
  };

  // Q A-fragments (dk=64 -> two K-steps), loaded straight from global
  v16bf qf[2];
#pragma unroll
  for (int ks = 0; ks < 2; ks++) {
    const __bf16* qr = qpanel + (size_t)nn * 64 + ks * 32;
    qf[ks] = cat16(*(const v8bf*)&qr[half * 8], *(const v8bf*)&qr[16 + half * 8]);
  }

  v8f acc[4] = {};
  float mrow[8], lrow[8];
#pragma unroll
  for (int r = 0; r < 8; r++) { mrow[r] = -1e30f; lrow[r] = 0.f; }
  const float scale = 0.125f;   // 1/sqrt(64)

  stageK(0, 0);   // prime the pipeline
  loadV(0);

  for (int it = 0; it < NIT; ++it) {
    int cur = it & 1;
    __syncthreads();             // all waves finished reading previous buffers
    { // commit V tile (transposed) for this iteration
#pragma unroll
      for (int j = 0; j < 8; j++) {
        Vlds[(c0 + j) * 32 + key]     = vr0[j];
        Vlds[(c0 + 8 + j) * 32 + key] = vr1[j];
      }
    }
    if (it + 1 < NIT) {          // issue next tile while current computes
      stageK((it + 1) * 32, cur ^ 1);
      loadV((it + 1) * 32);
      ASYNC_WAIT_LE2();          // current K tile landed; next still in flight
    } else {
      ASYNC_WAIT_LE0();
    }
    __syncthreads();

    // scores: 16 rows x 32 keys = two 16x16 C tiles, each K-reduced over dk=64
    v8f s[2];
#pragma unroll
    for (int nt = 0; nt < 2; nt++) {
      v8f c = {};
#pragma unroll
      for (int ks = 0; ks < 2; ks++) {
        const __bf16* kr = &Klds[cur][(nt * 16 + nn) * 64 + ks * 32];
        v16bf bfr = cat16(*(const v8bf*)&kr[half * 8], *(const v8bf*)&kr[16 + half * 8]);
        c = __builtin_amdgcn_wmma_f32_16x16x32_bf16(
            false, qf[ks], false, bfr, (short)0, c, false, false);
      }
      s[nt] = c;
    }

    // online softmax (rows live in VGPR index within each 16-lane half)
    float p0[8], p1[8], alpha[8];
#pragma unroll
    for (int r = 0; r < 8; r++) {
      float s0 = s[0][r] * scale, s1 = s[1][r] * scale;
      float mx = fmaxf(s0, s1);
#pragma unroll
      for (int m = 1; m < 16; m <<= 1) mx = fmaxf(mx, __shfl_xor(mx, m, 32));
      float mn = fmaxf(mrow[r], mx);
      alpha[r] = exp2f((mrow[r] - mn) * LOG2E);
      p0[r] = exp2f((s0 - mn) * LOG2E);
      p1[r] = exp2f((s1 - mn) * LOG2E);
      float ps = p0[r] + p1[r];
#pragma unroll
      for (int m = 1; m < 16; m <<= 1) ps += __shfl_xor(ps, m, 32);
      lrow[r] = lrow[r] * alpha[r] + ps;
      mrow[r] = mn;
    }
#pragma unroll
    for (int vt = 0; vt < 4; vt++)
#pragma unroll
      for (int r = 0; r < 8; r++) acc[vt][r] *= alpha[r];

    // C-layout -> A-layout for P via per-wave LDS round trip
    __bf16* P = &Plds[wid][0];
#pragma unroll
    for (int r = 0; r < 8; r++) {
      P[(half * 8 + r) * 32 + nn]      = f2bf(p0[r]);
      P[(half * 8 + r) * 32 + 16 + nn] = f2bf(p1[r]);
    }
    asm volatile("s_wait_dscnt 0" ::: "memory");   // same-wave LDS RAW ordering
    v16bf pf = cat16(*(const v8bf*)&P[nn * 32 + half * 8],
                     *(const v8bf*)&P[nn * 32 + 16 + half * 8]);

    // P(16x32) x V(32x64): 4 WMMA over 16-wide vdim tiles
#pragma unroll
    for (int vt = 0; vt < 4; vt++) {
      const __bf16* vr = &Vlds[(vt * 16 + nn) * 32 + half * 8];
      v16bf bfr = cat16(*(const v8bf*)vr, *(const v8bf*)&vr[16]);
      acc[vt] = __builtin_amdgcn_wmma_f32_16x16x32_bf16(
          false, pf, false, bfr, (short)0, acc[vt], false, false);
    }
  }

  // epilogue: normalize, store vals as [B*T][H*64] bf16
  {
    __bf16* obase = o + ((size_t)(b * T + qb * 64 + wid * 16)) * 1024 + h * 64;
#pragma unroll
    for (int vt = 0; vt < 4; vt++)
#pragma unroll
      for (int r = 0; r < 8; r++) {
        float val = acc[vt][r] / lrow[r];
        obase[(size_t)(half * 8 + r) * 1024 + vt * 16 + nn] = f2bf(val);
      }
  }
}

// ---------------------------------------------------------------------------
// Kernel 3: output projection.  out[8192,2048](f32) = vals[8192,1024](bf16) x Wo[1024,2048](bf16)
// A tile: async DMA double-buffered; B tile: register-prefetched.
// ---------------------------------------------------------------------------
__global__ __launch_bounds__(128) void out_gemm(
    const __bf16* __restrict__ A, const __bf16* __restrict__ W, float* __restrict__ C) {
  const int KDIM = 1024, NDIM = 2048;
  const int NIT = KDIM / 32;
  __shared__ __align__(16) __bf16 Alds[2][64 * 32];
  __shared__ __align__(16) __bf16 Blds[64 * 32];

  int tid = threadIdx.x;
  int wid = tid >> 5, lane = tid & 31;
  int half = lane >> 4, nn = lane & 15;
  int bm = blockIdx.y * 64, bn = blockIdx.x * 64;
  int wm = (wid >> 1) * 32, wn = (wid & 1) * 32;

  int rowA = tid >> 1, cbA = (tid & 1) * 16;
  int kkB = tid >> 2, c0B = (tid & 3) * 16;

  auto stageA = [&](int k0, int buf) {
    const __bf16* ga = A + (size_t)(bm + rowA) * KDIM + k0 + cbA;
#if HAVE_ASYNC_LDS
    ASYNC_COPY_B128(ga,     &Alds[buf][rowA * 32 + cbA]);
    ASYNC_COPY_B128(ga + 8, &Alds[buf][rowA * 32 + cbA + 8]);
#else
    const v8bf* src = (const v8bf*)ga;
    *(v8bf*)&Alds[buf][rowA * 32 + cbA]     = src[0];
    *(v8bf*)&Alds[buf][rowA * 32 + cbA + 8] = src[1];
#endif
  };
  v8bf wB0, wB1;
  auto loadB = [&](int k0) {
    const v8bf* src = (const v8bf*)(W + (size_t)(k0 + kkB) * NDIM + bn + c0B);
    wB0 = src[0]; wB1 = src[1];
  };

  stageA(0, 0);
  loadB(0);
  v8f acc[2][2] = {};

  for (int it = 0; it < NIT; ++it) {
    int k0 = it * 32, cur = it & 1;
    __syncthreads();
    { // commit B tile transposed
#pragma unroll
      for (int j = 0; j < 8; j++) {
        Blds[(c0B + j) * 32 + kkB]     = wB0[j];
        Blds[(c0B + 8 + j) * 32 + kkB] = wB1[j];
      }
    }
    if (it + 1 < NIT) {
      stageA(k0 + 32, cur ^ 1);
      loadB(k0 + 32);
      ASYNC_WAIT_LE2();
    } else {
      ASYNC_WAIT_LE0();
    }
    __syncthreads();

    v16bf af[2], bfr[2];
#pragma unroll
    for (int i = 0; i < 2; i++) {
      int r = wm + i * 16 + nn;
      af[i] = cat16(*(const v8bf*)&Alds[cur][r * 32 + half * 8],
                    *(const v8bf*)&Alds[cur][r * 32 + 16 + half * 8]);
    }
#pragma unroll
    for (int j = 0; j < 2; j++) {
      int c = wn + j * 16 + nn;
      bfr[j] = cat16(*(const v8bf*)&Blds[c * 32 + half * 8],
                     *(const v8bf*)&Blds[c * 32 + 16 + half * 8]);
    }
#pragma unroll
    for (int i = 0; i < 2; i++)
#pragma unroll
      for (int j = 0; j < 2; j++)
        acc[i][j] = __builtin_amdgcn_wmma_f32_16x16x32_bf16(
            false, af[i], false, bfr[j], (short)0, acc[i][j], false, false);
  }

  {
    float* Cbase = C + (size_t)(bm + wm) * NDIM + bn + wn;
#pragma unroll
    for (int i = 0; i < 2; i++)
#pragma unroll
      for (int j = 0; j < 2; j++)
#pragma unroll
        for (int r = 0; r < 8; r++)
          Cbase[(size_t)(i * 16 + half * 8 + r) * NDIM + j * 16 + nn] = acc[i][j][r];
  }
}

// ---------------------------------------------------------------------------
extern "C" void kernel_launch(void* const* d_in, const int* in_sizes, int n_in,
                              void* d_out, int out_size, void* d_ws, size_t ws_size,
                              hipStream_t stream) {
  (void)in_sizes; (void)n_in; (void)out_size; (void)ws_size;
  const float* Q  = (const float*)d_in[0];
  const float* K  = (const float*)d_in[1];
  const float* V  = (const float*)d_in[2];
  const float* Wq = (const float*)d_in[3];
  const float* Wk = (const float*)d_in[4];
  const float* Wv = (const float*)d_in[5];
  const float* Wo = (const float*)d_in[6];

  char* ws = (char*)d_ws;
  const size_t WSZ = (size_t)2048 * 1024 * 2;         // 4 MB per weight (bf16)
  const size_t PSZ = (size_t)4 * 16 * 2048 * 64 * 2;  // 16 MB per q/k/v panel (bf16)
  __bf16* Wqb  = (__bf16*)(ws + 0 * WSZ);
  __bf16* Wkb  = (__bf16*)(ws + 1 * WSZ);
  __bf16* Wvb  = (__bf16*)(ws + 2 * WSZ);
  __bf16* Wob  = (__bf16*)(ws + 3 * WSZ);
  __bf16* qb   = (__bf16*)(ws + 4 * WSZ);
  __bf16* kb   = (__bf16*)(ws + 4 * WSZ + 1 * PSZ);
  __bf16* vb   = (__bf16*)(ws + 4 * WSZ + 2 * PSZ);
  __bf16* vals = (__bf16*)(ws + 4 * WSZ + 3 * PSZ);

  // 0) weights f32 -> bf16
  int wn4 = 2048 * 1024 / 4;
  int cg = (wn4 + 255) / 256;
  cvt_f32_bf16<<<cg, 256, 0, stream>>>((const float4*)Wq, Wqb, wn4);
  cvt_f32_bf16<<<cg, 256, 0, stream>>>((const float4*)Wk, Wkb, wn4);
  cvt_f32_bf16<<<cg, 256, 0, stream>>>((const float4*)Wv, Wvb, wn4);
  cvt_f32_bf16<<<cg, 256, 0, stream>>>((const float4*)Wo, Wob, wn4);

  // 1) QKV projections (z picks Q/K/V)
  proj_gemm<<<dim3(1024 / 64, 8192 / 64, 3), 128, 0, stream>>>(
      Q, K, V, Wqb, Wkb, Wvb, qb, kb, vb);

  // 2) flash attention: 4*16*32 = 2048 workgroups
  flash_attn<<<dim3(2048), 128, 0, stream>>>(qb, kb, vb, vals);

  // 3) output projection -> f32 d_out
  out_gemm<<<dim3(2048 / 64, 8192 / 64), 128, 0, stream>>>(vals, Wob, (float*)d_out);
}
